// DirichletMoE_83949430768026
// MI455X (gfx1250) — compile-verified, matrix operand order checked by reading, not run on hardware
//
#include <hip/hip_runtime.h>
#include <hip/hip_bf16.h>
#include <math.h>

// ---------------------------------------------------------------------------
// DirichletMoE for MI455X (gfx1250, wave32, WMMA bf16 16x16x32)
// Top-2 sparse dispatch: only selected experts computed (exact vs reference).
// TM=64 token tiles, 8 waves, 256KB LDS, async global->LDS staging.
// ---------------------------------------------------------------------------

typedef __bf16 bf16_t;
typedef __attribute__((ext_vector_type(16))) __bf16 v16bf;
typedef __attribute__((ext_vector_type(8)))  float  v8f;

#define B_   8192
#define IN_  512
#define HID_ 1024
#define OUT_ 256
#define E_   16
#define TM   64      // token tile (M) per block

union Frag { v16bf v; unsigned int u[8]; };

// 16-bit A-matrix 16x32 layout (ISA 7.12.2): lanes 0-15 rows M=0..15 hold
// K={0..7,16..23}; lanes 16-31 hold K={8..15,24..31}. B mirrors with N=lane%16.
__device__ __forceinline__ void ld_frag(Frag& f, const bf16_t* base, int ld,
                                        int row, int kk, int lane) {
  const int r  = row + (lane & 15);
  const int kb = kk + ((lane >> 4) << 3);
  const unsigned int* p0 = (const unsigned int*)(base + (size_t)r * ld + kb);
  const unsigned int* p1 = (const unsigned int*)(base + (size_t)r * ld + kb + 16);
  f.u[0] = p0[0]; f.u[1] = p0[1]; f.u[2] = p0[2]; f.u[3] = p0[3];
  f.u[4] = p1[0]; f.u[5] = p1[1]; f.u[6] = p1[2]; f.u[7] = p1[3];
}

__device__ __forceinline__ v8f wmma_bf16(const Frag& a, const Frag& b, v8f c) {
  return __builtin_amdgcn_wmma_f32_16x16x32_bf16(
      /*neg_a=*/false, a.v, /*neg_b=*/false, b.v,
      /*c_mod=*/(short)0, c, /*reuse_a=*/false, /*reuse_b=*/false);
}

// One block = 8 waves; wave (m_tile = wid&3, n_half = wid>>2) computes a
// 16 x (NTOT/2) strip in 64-wide N chunks (4 accumulators of 8 VGPRs each).
template <int KTOT, int NTOT, bool RELU, bool OUT_F32>
__device__ __forceinline__ void gemm_tile(const bf16_t* __restrict__ A, int lda,
                                          const bf16_t* __restrict__ Bt, int ldb,
                                          const float* __restrict__ bias,
                                          bf16_t* __restrict__ outB,
                                          float* __restrict__ outF, int ldo,
                                          int wid, int lane) {
  const int mrow   = (wid & 3) * 16;
  const int nStart = (wid >> 2) * (NTOT / 2);
  for (int nc = nStart; nc < nStart + NTOT / 2; nc += 64) {
    v8f acc[4];
#pragma unroll
    for (int t = 0; t < 4; ++t)
#pragma unroll
      for (int r = 0; r < 8; ++r) acc[t][r] = 0.f;

    for (int kk = 0; kk < KTOT; kk += 32) {
      Frag a;
      ld_frag(a, A, lda, mrow, kk, lane);
      if (kk + 32 < KTOT)  // keep next weight chunk warm (near caches)
        __builtin_prefetch(Bt + (size_t)(nc + (lane & 15)) * ldb + kk + 32, 0, 3);
#pragma unroll
      for (int t = 0; t < 4; ++t) {
        Frag b;
        ld_frag(b, Bt, ldb, nc + t * 16, kk, lane);
        acc[t] = wmma_bf16(a, b, acc[t]);
      }
    }
    // C/D layout: VGPR r -> M = mrow + r + (lane>=16 ? 8 : 0), N = nc+t*16+(lane&15)
    const int mo = mrow + ((lane >> 4) << 3);
    const int no = lane & 15;
#pragma unroll
    for (int t = 0; t < 4; ++t) {
      const int n = nc + t * 16 + no;
      const float bz = bias[n];
#pragma unroll
      for (int r = 0; r < 8; ++r) {
        float v = acc[t][r] + bz;
        if (RELU) v = v > 0.f ? v : 0.f;
        const int m = mo + r;
        if (OUT_F32) outF[m * ldo + n] = v;
        else         outB[m * ldo + n] = (bf16_t)v;
      }
    }
  }
}

// ---------------------------------------------------------------------------
// Prep kernels
// ---------------------------------------------------------------------------
extern "C" __global__ void cast_x_kernel(const float* __restrict__ x,
                                         bf16_t* __restrict__ xb, int n) {
  for (int i = blockIdx.x * blockDim.x + threadIdx.x; i < n;
       i += gridDim.x * blockDim.x)
    xb[i] = (bf16_t)x[i];
}

// dst[e][c][r] = (bf16) src[e][r][c]   (N-major weights -> K-contiguous B rows)
extern "C" __global__ void tcast_kernel(const float* __restrict__ src,
                                        bf16_t* __restrict__ dst,
                                        int R, int C, int total) {
  for (int i = blockIdx.x * blockDim.x + threadIdx.x; i < total;
       i += gridDim.x * blockDim.x) {
    const int r = i % R;
    const int t = i / R;
    const int c = t % C;
    const int e = t / C;
    dst[i] = (bf16_t)src[(size_t)e * R * C + (size_t)r * C + c];
  }
}

extern "C" __global__ void init_kernel(float* __restrict__ outf, int n_zero,
                                       int* __restrict__ counts,
                                       float* __restrict__ importance,
                                       float* __restrict__ loadv) {
  const int i = blockIdx.x * blockDim.x + threadIdx.x;
  for (int j = i; j < n_zero; j += gridDim.x * blockDim.x) outf[j] = 0.f;
  if (i < E_) { counts[i] = 0; importance[i] = 0.f; loadv[i] = 0.f; }
}

// ---------------------------------------------------------------------------
// Noisy top-2 gating: one wave per token. Lanes 0-15: clean logits (e=lane),
// lanes 16-31: noise pre-activation. Shuffle-broadcast for top-2 selection.
// ---------------------------------------------------------------------------
extern "C" __global__ void __launch_bounds__(128)
gating_kernel(const float* __restrict__ x, const float* __restrict__ noise,
              const float* __restrict__ gW, const float* __restrict__ gb,
              const float* __restrict__ nW, const float* __restrict__ nb,
              int* __restrict__ counts, int* __restrict__ assign,
              float* __restrict__ gateval, float* __restrict__ importance,
              float* __restrict__ loadv) {
  const int lane = threadIdx.x & 31;
  const int wid  = threadIdx.x >> 5;
  const int tok  = blockIdx.x * 4 + wid;
  const int e    = lane & 15;
  const bool np  = lane >= 16;

  const float* W = np ? nW : gW;
  float acc = (np ? nb : gb)[e];
  const float* xr = x + (size_t)tok * IN_;
  for (int i = 0; i < IN_; ++i) acc += xr[i] * W[i * E_ + e];

  const float pre_noise = __shfl(acc, e + 16);
  const float sp = (pre_noise > 20.f) ? pre_noise : log1pf(__expf(pre_noise));
  const float noisy = acc + noise[(size_t)tok * E_ + e] * sp;  // lanes 0-15

  float best = -1e30f, second = -1e30f;
  int bi = 0, si = 0;
  for (int q = 0; q < 16; ++q) {
    const float v = __shfl(noisy, q);
    if (v > best)        { second = best; si = bi; best = v; bi = q; }
    else if (v > second) { second = v; si = q; }
  }
  if (lane == 0) {
    const float e1 = __expf(second - best);
    const float g0 = 1.f / (1.f + e1);
    const float g1 = e1 / (1.f + e1);
    int s0 = atomicAdd(&counts[bi], 1);
    assign[bi * B_ + s0] = tok;  gateval[bi * B_ + s0] = g0;
    atomicAdd(&importance[bi], g0); atomicAdd(&loadv[bi], 1.f);
    int s1 = atomicAdd(&counts[si], 1);
    assign[si * B_ + s1] = tok;  gateval[si * B_ + s1] = g1;
    atomicAdd(&importance[si], g1); atomicAdd(&loadv[si], 1.f);
  }
}

// ---------------------------------------------------------------------------
// Fused expert pipeline: x->h1->h2->(softmax p, alpha0)->scatter combine.
// grid = (E, B/TM), block = 256 (8 waves), 256 KB dynamic LDS.
// ---------------------------------------------------------------------------
extern "C" __global__ void __launch_bounds__(256)
expert_kernel(const bf16_t* __restrict__ xb,
              const bf16_t* __restrict__ W1t, const float* __restrict__ b1,
              const bf16_t* __restrict__ W2t, const float* __restrict__ b2,
              const bf16_t* __restrict__ Wpt, const float* __restrict__ bp,
              const float* __restrict__ Wa, const float* __restrict__ ba,
              const int* __restrict__ counts, const int* __restrict__ assign,
              const float* __restrict__ gateval,
              float* __restrict__ p_hat, float* __restrict__ alpha0) {
  extern __shared__ char smem[];
  bf16_t* bufA = (bf16_t*)smem;                  // 128KB: x tile (64x512) then h2 (64x1024)
  bf16_t* bufB = (bf16_t*)(smem + 128 * 1024);   // 128KB: h1 (64x1024)
  float*  lgts = (float*)(smem + 128 * 1024);    //  64KB overlay on bufB: logits (64x256)

  const int e   = blockIdx.x;
  const int cnt = counts[e];
  const int t0  = blockIdx.y * TM;
  if (t0 >= cnt) return;

  const int tid = threadIdx.x, lane = tid & 31, wid = tid >> 5;

  // Stage gathered x rows (bf16) straight into LDS via CDNA5 async copies:
  // 4 threads per row, each moves 256B as 16x b128. INST_OFFSET applies to
  // both the LDS and global address (ISA 10.7 async pseudocode), so base
  // addresses stay fixed and only offset: steps.
  {
    const int row = tid >> 2, q = tid & 3;
    int slot = t0 + row;
    if (slot >= cnt) slot = cnt - 1;
    const int tok = assign[e * B_ + slot];
    const bf16_t* gsrc = xb + (size_t)tok * IN_ + q * 128;        // 256B chunk
    const unsigned ldst = (unsigned)(uintptr_t)(bufA + row * IN_ + q * 128);
#pragma unroll
    for (int i = 0; i < 16; ++i) {
      asm volatile("global_load_async_to_lds_b128 %0, %1, off offset:%2"
                   :: "v"(ldst), "v"(gsrc), "i"(i * 16) : "memory");
    }
    asm volatile("s_wait_asynccnt 0x0" ::: "memory");
  }
  __syncthreads();

  // h1 = relu(x @ W1 + b1)
  gemm_tile<IN_, HID_, true, false>(bufA, IN_, W1t + (size_t)e * HID_ * IN_, IN_,
                                    b1 + e * HID_, bufB, nullptr, HID_, wid, lane);
  __syncthreads();
  // h2 = relu(h1 @ W2 + b2)  (overwrites x region)
  gemm_tile<HID_, HID_, true, false>(bufB, HID_, W2t + (size_t)e * HID_ * HID_, HID_,
                                     b2 + e * HID_, bufA, nullptr, HID_, wid, lane);
  __syncthreads();
  // logits = h2 @ Wp + bp (float, overlays h1)
  gemm_tile<HID_, OUT_, false, true>(bufA, HID_, Wpt + (size_t)e * OUT_ * HID_, HID_,
                                     bp + e * OUT_, nullptr, lgts, OUT_, wid, lane);
  __syncthreads();

  // Per-row epilogue: alpha0 dot + softmax + gated atomic scatter.
  const float* Wag = Wa + e * HID_;
  const float  bag = ba[e];
#pragma unroll
  for (int rr = 0; rr < 8; ++rr) {
    const int m = wid * 8 + rr;
    const int slot = t0 + m;

    float s = 0.f;
    for (int k = lane; k < HID_; k += 32) s += (float)bufA[m * HID_ + k] * Wag[k];
#pragma unroll
    for (int off = 16; off > 0; off >>= 1) s += __shfl_xor(s, off);

    float mx = -1e30f, v[8];
#pragma unroll
    for (int j = 0; j < 8; ++j) {
      v[j] = lgts[m * OUT_ + lane + j * 32];
      mx = fmaxf(mx, v[j]);
    }
#pragma unroll
    for (int off = 16; off > 0; off >>= 1) mx = fmaxf(mx, __shfl_xor(mx, off));
    float sm = 0.f;
#pragma unroll
    for (int j = 0; j < 8; ++j) { v[j] = __expf(v[j] - mx); sm += v[j]; }
#pragma unroll
    for (int off = 16; off > 0; off >>= 1) sm += __shfl_xor(sm, off);

    if (slot < cnt) {
      const int   tok = assign[e * B_ + slot];
      const float g   = gateval[e * B_ + slot];
      if (lane == 0) {
        const float pre = s + bag;
        const float spv = (pre > 20.f) ? pre : log1pf(__expf(pre));
        const float a0v = fminf(fmaxf(spv + 10.f, 1.f), 500.f);
        atomicAdd(&alpha0[tok], g * a0v);
      }
      const float inv = g / sm;
#pragma unroll
      for (int j = 0; j < 8; ++j)
        atomicAdd(&p_hat[(size_t)tok * OUT_ + lane + j * 32], v[j] * inv);
    }
  }
}

// ---------------------------------------------------------------------------
// Post: aux loss (ddof=1 std) and p_hat row normalization.
// ---------------------------------------------------------------------------
extern "C" __global__ void aux_kernel(const float* __restrict__ importance,
                                      const float* __restrict__ loadv,
                                      float* __restrict__ out_aux) {
  if (threadIdx.x == 0 && blockIdx.x == 0) {
    float mi = 0.f, ml = 0.f;
    for (int i = 0; i < E_; ++i) { mi += importance[i]; ml += loadv[i]; }
    mi /= (float)E_; ml /= (float)E_;
    float vi = 0.f, vl = 0.f;
    for (int i = 0; i < E_; ++i) {
      const float di = importance[i] - mi; vi += di * di;
      const float dl = loadv[i] - ml;      vl += dl * dl;
    }
    vi /= (float)(E_ - 1); vl /= (float)(E_ - 1);
    *out_aux = sqrtf(vi) / (mi + 1e-8f) + sqrtf(vl) / (ml + 1e-8f);
  }
}

extern "C" __global__ void __launch_bounds__(256)
norm_kernel(float* __restrict__ p_hat) {
  __shared__ float part[8];
  const int row = blockIdx.x, tid = threadIdx.x, lane = tid & 31, wid = tid >> 5;
  const float v = p_hat[(size_t)row * OUT_ + tid];
  float s = v;
#pragma unroll
  for (int off = 16; off > 0; off >>= 1) s += __shfl_xor(s, off);
  if (lane == 0) part[wid] = s;
  __syncthreads();
  float tot = 0.f;
#pragma unroll
  for (int w = 0; w < 8; ++w) tot += part[w];
  p_hat[(size_t)row * OUT_ + tid] = v / (tot + 1e-8f);
}

// ---------------------------------------------------------------------------
// Host launch
// ---------------------------------------------------------------------------
extern "C" void kernel_launch(void* const* d_in, const int* in_sizes, int n_in,
                              void* d_out, int out_size, void* d_ws, size_t ws_size,
                              hipStream_t stream) {
  const float* x   = (const float*)d_in[0];
  const float* noi = (const float*)d_in[1];
  const float* gW  = (const float*)d_in[2];
  const float* gb  = (const float*)d_in[3];
  const float* nW  = (const float*)d_in[4];
  const float* nb  = (const float*)d_in[5];
  const float* W1  = (const float*)d_in[6];
  const float* b1  = (const float*)d_in[7];
  const float* W2  = (const float*)d_in[8];
  const float* b2  = (const float*)d_in[9];
  const float* Wp  = (const float*)d_in[10];
  const float* bp  = (const float*)d_in[11];
  const float* Wa  = (const float*)d_in[12];
  const float* ba  = (const float*)d_in[13];

  float* out    = (float*)d_out;
  float* p_hat  = out;                           // [B, OUT]
  float* alpha0 = out + (size_t)B_ * OUT_;       // [B]
  float* auxp   = out + (size_t)B_ * OUT_ + B_;  // scalar

  char* w = (char*)d_ws;
  bf16_t* xb  = (bf16_t*)w;  w += (size_t)B_ * IN_ * 2;         // 8 MB
  bf16_t* W1t = (bf16_t*)w;  w += (size_t)E_ * HID_ * IN_ * 2;  // 16 MB
  bf16_t* W2t = (bf16_t*)w;  w += (size_t)E_ * HID_ * HID_ * 2; // 32 MB
  bf16_t* Wpt = (bf16_t*)w;  w += (size_t)E_ * OUT_ * HID_ * 2; // 8 MB
  int*    counts     = (int*)w;   w += 256;
  int*    assign     = (int*)w;   w += (size_t)E_ * B_ * 4;     // 512 KB
  float*  gateval    = (float*)w; w += (size_t)E_ * B_ * 4;     // 512 KB
  float*  importance = (float*)w; w += 256;
  float*  loadv      = (float*)w; w += 256;

  init_kernel<<<2048, 256, 0, stream>>>(p_hat, B_ * OUT_ + B_, counts, importance, loadv);
  cast_x_kernel<<<2048, 256, 0, stream>>>(x, xb, B_ * IN_);
  tcast_kernel<<<4096, 256, 0, stream>>>(W1, W1t, IN_, HID_, E_ * IN_ * HID_);
  tcast_kernel<<<8192, 256, 0, stream>>>(W2, W2t, HID_, HID_, E_ * HID_ * HID_);
  tcast_kernel<<<2048, 256, 0, stream>>>(Wp, Wpt, HID_, OUT_, E_ * HID_ * OUT_);

  gating_kernel<<<B_ / 4, 128, 0, stream>>>(x, noi, gW, gb, nW, nb,
                                            counts, assign, gateval, importance, loadv);

  dim3 eg(E_, B_ / TM);
  expert_kernel<<<eg, 256, 256 * 1024, stream>>>(xb, W1t, b1, W2t, b2, Wpt, bp,
                                                 Wa, ba, counts, assign, gateval,
                                                 p_hat, alpha0);

  aux_kernel<<<1, 32, 0, stream>>>(importance, loadv, auxp);
  norm_kernel<<<B_, 256, 0, stream>>>(p_hat);
}
